// MultiParamAttentionInterpolator_64922725646862
// MI455X (gfx1250) — compile-verified, matrix operand order checked by reading, not run on hardware
//
#include <hip/hip_runtime.h>

#define N_  16
#define T_  64
#define GX_ 256
#define GY_ 256
#define GQ_ 256
#define H_  4
#define D_  8

typedef __attribute__((ext_vector_type(16))) _Float16 v16h;
typedef __attribute__((ext_vector_type(8)))  float    v8f;

// ---- gfx1250 async global->LDS path (probe-gated, with sync fallback) ------
#if defined(__has_builtin)
#  if __has_builtin(__builtin_amdgcn_global_load_async_to_lds_b32)
#    define HAVE_ASYNC_LDS 1
#  endif
#  if __has_builtin(__builtin_amdgcn_global_load_async_to_lds_b128)
#    define HAVE_ASYNC_LDS_B128 1
#  endif
#endif

__device__ __forceinline__ void wait_async0() {
#if defined(__has_builtin) && __has_builtin(__builtin_amdgcn_s_wait_asynccnt)
    __builtin_amdgcn_s_wait_asynccnt(0);
#else
    asm volatile("s_wait_asynccnt 0" ::: "memory");
#endif
}

#ifdef HAVE_ASYNC_LDS
typedef __attribute__((address_space(1))) int  gint_t;
typedef __attribute__((address_space(3))) int  lint_t;
__device__ __forceinline__ void async_cp_b32(float* ldst, const float* gsrc) {
    __builtin_amdgcn_global_load_async_to_lds_b32(
        (gint_t*)(gsrc), (lint_t*)(ldst), 0, 0);
}
#ifdef HAVE_ASYNC_LDS_B128
typedef __attribute__((ext_vector_type(4))) int v4i_;
typedef __attribute__((address_space(1))) v4i_ gv4i_t;
typedef __attribute__((address_space(3))) v4i_ lv4i_t;
__device__ __forceinline__ void async_cp_b128(float* ldst, const float* gsrc) {
    __builtin_amdgcn_global_load_async_to_lds_b128(
        (gv4i_t*)(gsrc), (lv4i_t*)(ldst), 0, 0);
}
#endif

// Stage the 4 source rows (c1:ix,ix+1 ; c2:ix,ix+1) of slice n into buf.
// b128: thread j copies 16B segment (j&63) of row (j>>6)  -> 1 instr/thread.
// b32 : thread j copies element j of each of the 4 rows   -> 4 instr/thread.
__device__ __forceinline__ void stage_slice(float (*b)[GQ_],
                                            const float* r1, const float* r2,
                                            int j) {
#ifdef HAVE_ASYNC_LDS_B128
    const int row = j >> 6;          // 0..3
    const int seg = (j & 63) << 2;   // float offset, 16B-aligned
    const float* src = (row == 0) ? (r1 + seg)
                     : (row == 1) ? (r1 + GY_ + seg)
                     : (row == 2) ? (r2 + seg)
                     :              (r2 + GY_ + seg);
    async_cp_b128(&b[row][seg], src);
#else
    async_cp_b32(&b[0][j], r1 + j);
    async_cp_b32(&b[1][j], r1 + GY_ + j);
    async_cp_b32(&b[2][j], r2 + j);
    async_cp_b32(&b[3][j], r2 + GY_ + j);
#endif
}
#endif // HAVE_ASYNC_LDS

// ---------------------------------------------------------------------------
// Kernel 1: precompute interpolation indices/fractions and the blend weights w.
// One block of 256 threads. Wave 0 computes the attention scores with a single
// V_WMMA_F32_16X16X32_F16 (attn = p @ A2 + b2, with q folded into A2/b2).
// ---------------------------------------------------------------------------
__global__ __launch_bounds__(256) void prep_weights_grids(
    const float* __restrict__ xg_g,
    const float* __restrict__ yg_g,
    const float* __restrict__ params,
    const float* __restrict__ Wq_w, const float* __restrict__ Wq_b,
    const float* __restrict__ Wk_w, const float* __restrict__ Wk_b,
    const float* __restrict__ lyt_p,
    const float* __restrict__ ccu_p,
    const float* __restrict__ cni_p,
    float* __restrict__ ws_w,
    int*   __restrict__ ws_ix, float* __restrict__ ws_fx,
    int*   __restrict__ ws_iy, float* __restrict__ ws_fy,
    float* __restrict__ out_w)
{
    __shared__ float s_xg[GX_];
    __shared__ float s_ys[GY_];
    __shared__ float s_attn[H_ * N_];

    const int tid = threadIdx.x;
    const float lyt = lyt_p[0];

    s_xg[tid] = xg_g[tid];
    s_ys[tid] = yg_g[tid] * (lyt / params[0]);
    __syncthreads();

    const float Lx = s_xg[GX_ - 1];
    const float u  = (float)tid * (1.0f / (float)(GQ_ - 1));

    {   // x axis searchsorted(side='right')
        const float qx = u * Lx;
        int lo = 0, hi = GX_;
        while (lo < hi) { int mid = (lo + hi) >> 1; if (s_xg[mid] <= qx) lo = mid + 1; else hi = mid; }
        int ix = lo - 1; ix = ix < 0 ? 0 : (ix > GX_ - 2 ? GX_ - 2 : ix);
        const float x0 = s_xg[ix], x1 = s_xg[ix + 1];
        ws_ix[tid] = ix;
        ws_fx[tid] = (qx - x0) / (x1 - x0);
    }
    {   // y axis
        const float qy = u * lyt;
        int lo = 0, hi = GY_;
        while (lo < hi) { int mid = (lo + hi) >> 1; if (s_ys[mid] <= qy) lo = mid + 1; else hi = mid; }
        int iy = lo - 1; iy = iy < 0 ? 0 : (iy > GY_ - 2 ? GY_ - 2 : iy);
        const float y0 = s_ys[iy], y1 = s_ys[iy + 1];
        ws_iy[tid] = iy;
        ws_fy[tid] = (qy - y0) / (y1 - y0);
    }
    __syncthreads();

    const float t0 = (lyt - 30.0f) * (1.0f / 90.0f);
    const float t1 = ccu_p[0] * (1.0f / 0.0029f);
    const float t2 = cni_p[0] * (1.0f / 0.0018f);

    if (tid < 32) {
        const int lane = tid;
        const int m = lane & 15;

        const float p0 = (params[m * 3 + 0] - 30.0f) * (1.0f / 90.0f);
        const float p1 = params[m * 3 + 1] * (1.0f / 0.0029f);
        const float p2 = params[m * 3 + 2] * (1.0f / 0.0018f);

        v16h A = {};
        v16h B = {};
        if (lane < 16) {
            A[0] = (_Float16)p0;
            A[1] = (_Float16)p1;
            A[2] = (_Float16)p2;
            A[3] = (_Float16)1.0f;
        }
        if (lane < H_) {
            const float inv_sqrt_d = 0.35355339059327373f;
            float a2_0 = 0.f, a2_1 = 0.f, a2_2 = 0.f, bb = 0.f;
            #pragma unroll
            for (int d = 0; d < D_; ++d) {
                const int c = lane * D_ + d;
                const float qv = Wq_b[c] + t0 * Wq_w[c * 3 + 0]
                                         + t1 * Wq_w[c * 3 + 1]
                                         + t2 * Wq_w[c * 3 + 2];
                a2_0 += Wk_w[c * 3 + 0] * qv;
                a2_1 += Wk_w[c * 3 + 1] * qv;
                a2_2 += Wk_w[c * 3 + 2] * qv;
                bb   += Wk_b[c] * qv;
            }
            B[0] = (_Float16)(a2_0 * inv_sqrt_d);
            B[1] = (_Float16)(a2_1 * inv_sqrt_d);
            B[2] = (_Float16)(a2_2 * inv_sqrt_d);
            B[3] = (_Float16)(bb   * inv_sqrt_d);
        }

        v8f C = {};
        C = __builtin_amdgcn_wmma_f32_16x16x32_f16(
                false, A, false, B, (short)0, C, false, false);

        const int col = lane & 15;
        if (col < H_) {
            #pragma unroll
            for (int r = 0; r < 8; ++r) {
                const int mm = (lane < 16) ? r : (r + 8);
                s_attn[col * N_ + mm] = C[r];
            }
        }
    }
    __syncthreads();

    if (tid == 0) {
        float attn_w[N_];
        #pragma unroll
        for (int n = 0; n < N_; ++n) attn_w[n] = 0.f;
        for (int h = 0; h < H_; ++h) {
            float mx = -1e30f;
            for (int n = 0; n < N_; ++n) mx = fmaxf(mx, s_attn[h * N_ + n]);
            float sum = 0.f;
            float e[N_];
            for (int n = 0; n < N_; ++n) { e[n] = __expf(s_attn[h * N_ + n] - mx); sum += e[n]; }
            const float inv = 1.0f / sum;
            for (int n = 0; n < N_; ++n) attn_w[n] += e[n] * inv;
        }
        float wv[N_];
        float ssum = 0.f;
        for (int n = 0; n < N_; ++n) {
            const float p0 = (params[n * 3 + 0] - 30.0f) * (1.0f / 90.0f);
            const float p1 = params[n * 3 + 1] * (1.0f / 0.0029f);
            const float p2 = params[n * 3 + 2] * (1.0f / 0.0018f);
            const float d0 = (p0 - t0) * 5.0f;
            const float d1 = (p1 - t1) * 5.0f;
            const float d2 = (p2 - t2) * 5.0f;
            wv[n] = __expf(-0.5f * (d0 * d0 + d1 * d1 + d2 * d2));
            ssum += wv[n];
        }
        float wsum = 0.f;
        for (int n = 0; n < N_; ++n) {
            wv[n] = (attn_w[n] * 0.25f) * (wv[n] / ssum);
            wsum += wv[n];
        }
        const float invw = 1.0f / wsum;
        for (int n = 0; n < N_; ++n) {
            const float wf = wv[n] * invw;
            ws_w[n]  = wf;
            out_w[n] = wf;
        }
    }
}

// ---------------------------------------------------------------------------
// Kernel 2: streaming weighted bilinear reduction.
// Async path: double-buffered GLOBAL_LOAD_ASYNC_TO_LDS_B128/B32 copies the
// four source rows for slice n+1 while slice n is blended from LDS. One
// barrier per iteration; ASYNCcnt hides the global latency.
// ---------------------------------------------------------------------------
__global__ __launch_bounds__(256) void interp_reduce(
    const float* __restrict__ c1, const float* __restrict__ c2,
    const float* __restrict__ ws_w,
    const int*   __restrict__ ws_ix, const float* __restrict__ ws_fx,
    const int*   __restrict__ ws_iy, const float* __restrict__ ws_fy,
    const float* __restrict__ ccu_p, const float* __restrict__ cni_p,
    float* __restrict__ out)
{
    __shared__ float wsh[N_];

    const int j  = threadIdx.x;
    const int t  = blockIdx.x >> 8;      // blockIdx.x = t*GQ + i
    const int ii = blockIdx.x & 255;

    if (j < N_) wsh[j] = ws_w[j];

    const int   ixv = ws_ix[ii];
    const float fxv = ws_fx[ii];
    const int   iyv = ws_iy[j];
    const float fyv = ws_fy[j];

    const size_t nstride = (size_t)T_ * GX_ * GY_;
    const size_t base    = ((size_t)t * GX_ + (size_t)ixv) * GY_;

    float acc1 = 0.f, acc2 = 0.f;

#ifdef HAVE_ASYNC_LDS
    // buf[s][row][j]: rows 0,1 = c1(ix,ix+1); rows 2,3 = c2(ix,ix+1)
    __shared__ float buf[2][4][GQ_];

    stage_slice(buf[0], c1 + base, c2 + base, j);   // prologue: slice 0

    #pragma unroll 2
    for (int n = 0; n < N_; ++n) {
        wait_async0();          // this wave's copies for slice n have landed
        __syncthreads();        // all waves' copies landed; buf[(n+1)&1] free

        if (n + 1 < N_) {       // overlap slice n+1 copies with slice n math
            const size_t nb = base + (size_t)(n + 1) * nstride;
            stage_slice(buf[(n + 1) & 1], c1 + nb, c2 + nb, j);
        }

        const float (*bc)[GQ_] = buf[n & 1];
        const float wn = wsh[n];

        const float a00 = bc[0][iyv], a01 = bc[0][iyv + 1];
        const float a10 = bc[1][iyv], a11 = bc[1][iyv + 1];
        const float e0  = __builtin_fmaf(fxv, a10 - a00, a00);
        const float e1  = __builtin_fmaf(fxv, a11 - a01, a01);
        acc1 = __builtin_fmaf(wn, __builtin_fmaf(fyv, e1 - e0, e0), acc1);

        const float b00 = bc[2][iyv], b01 = bc[2][iyv + 1];
        const float b10 = bc[3][iyv], b11 = bc[3][iyv + 1];
        const float f0  = __builtin_fmaf(fxv, b10 - b00, b00);
        const float f1  = __builtin_fmaf(fxv, b11 - b01, b01);
        acc2 = __builtin_fmaf(wn, __builtin_fmaf(fyv, f1 - f0, f0), acc2);
    }
#else
    // fallback: synchronous loads + x-lerp staging in LDS
    __shared__ float l1[GQ_];
    __shared__ float l2[GQ_];
    __syncthreads();

    #pragma unroll 4
    for (int n = 0; n < N_; ++n) {
        const float* r1 = c1 + base + (size_t)n * nstride;
        const float* r2 = c2 + base + (size_t)n * nstride;

        const float a0 = r1[j];
        const float a1 = r1[GY_ + j];
        const float b0 = r2[j];
        const float b1 = r2[GY_ + j];

        if (n + 1 < N_) {
            __builtin_prefetch(r1 + nstride + j, 0, 0);
            __builtin_prefetch(r2 + nstride + j, 0, 0);
        }

        l1[j] = __builtin_fmaf(fxv, a1 - a0, a0);
        l2[j] = __builtin_fmaf(fxv, b1 - b0, b0);
        __syncthreads();

        const float wn = wsh[n];
        const float u0 = l1[iyv], u1 = l1[iyv + 1];
        const float v0 = l2[iyv], v1 = l2[iyv + 1];
        acc1 = __builtin_fmaf(wn, __builtin_fmaf(fyv, u1 - u0, u0), acc1);
        acc2 = __builtin_fmaf(wn, __builtin_fmaf(fyv, v1 - v0, v0), acc2);
        __syncthreads();
    }
#endif

    const size_t o = ((size_t)t * GQ_ + (size_t)ii) * GQ_ + (size_t)j;
    out[o] = (j == 0) ? ccu_p[0] : acc1;                      // c1[:,:,0]  = c_cu
    out[(size_t)T_ * GQ_ * GQ_ + o] =
        (j == GQ_ - 1) ? cni_p[0] : acc2;                     // c2[:,:,-1] = c_ni
}

extern "C" void kernel_launch(void* const* d_in, const int* in_sizes, int n_in,
                              void* d_out, int out_size, void* d_ws, size_t ws_size,
                              hipStream_t stream) {
    const float* c1_src  = (const float*)d_in[0];
    const float* c2_src  = (const float*)d_in[1];
    const float* x_grids = (const float*)d_in[2];
    const float* y_grids = (const float*)d_in[3];
    const float* params  = (const float*)d_in[4];
    const float* Wq_w    = (const float*)d_in[5];
    const float* Wq_b    = (const float*)d_in[6];
    const float* Wk_w    = (const float*)d_in[7];
    const float* Wk_b    = (const float*)d_in[8];
    const float* lyt     = (const float*)d_in[9];
    const float* ccu     = (const float*)d_in[10];
    const float* cni     = (const float*)d_in[11];

    float* out = (float*)d_out;
    float* out_w = out + (size_t)2 * T_ * GQ_ * GQ_;

    float* wsf   = (float*)d_ws;
    float* ws_w  = wsf;
    int*   ws_ix = (int*)(wsf + 16);
    float* ws_fx = wsf + 16 + 256;
    int*   ws_iy = (int*)(wsf + 16 + 512);
    float* ws_fy = wsf + 16 + 768;

    prep_weights_grids<<<1, 256, 0, stream>>>(
        x_grids, y_grids, params, Wq_w, Wq_b, Wk_w, Wk_b,
        lyt, ccu, cni, ws_w, ws_ix, ws_fx, ws_iy, ws_fy, out_w);

    interp_reduce<<<T_ * GQ_, 256, 0, stream>>>(
        c1_src, c2_src, ws_w, ws_ix, ws_fx, ws_iy, ws_fy, ccu, cni, out);
}
